// Gaussian_Fusing_532575945370
// MI455X (gfx1250) — matrix operand order repaired and checked by reading, not compiled, not run on hardware
//
#include <hip/hip_runtime.h>

typedef __attribute__((ext_vector_type(16))) __bf16 v16bf;
typedef __attribute__((ext_vector_type(8)))  __bf16 v8bf;
typedef __attribute__((ext_vector_type(4)))  __bf16 v4bf;
typedef __attribute__((ext_vector_type(8)))  float  v8f;
typedef __attribute__((ext_vector_type(4)))  int    v4i;

#define DD 1024
#define SS 1024
#define BB 8
#define M_TOTAL (BB * SS)   // 8192 rows
#define TAPS 16

// Block tile: 128 (M) x 64 (N), 8 waves in 4(M) x 2(N); wave tile 32x32.
#define BM 128
#define BN 64
#define KC 32               // K per stage (one WMMA K-step)

// LDS layout per buffer:
//   [0,      8192)  : x A-tile, 128 rows x 64B (chunk-swizzled)
//   [8192,  16384)  : f A-tile, 128 rows x 64B (chunk-swizzled)
//   [16384, 28672)  : B frags, (mat*4 + nbLocal)*1024 + h2*512 + lane*16
#define LDS_A_X   0
#define LDS_A_F   8192
#define LDS_B     16384
#define LDS_BUF   28672

#define HAVE_ASYNC_LDS __has_builtin(__builtin_amdgcn_global_load_async_to_lds_b128)

__device__ __forceinline__ __bf16 f2bf(float f) {
  union { float f; unsigned u; } in; in.f = f;
  unsigned r = in.u + 0x7FFFu + ((in.u >> 16) & 1u);   // round-to-nearest-even
  union { unsigned short s; __bf16 h; } out; out.s = (unsigned short)(r >> 16);
  return out.h;
}

#if HAVE_ASYNC_LDS
__device__ __forceinline__ void async_cp16(const void* g, void* l) {
  __builtin_amdgcn_global_load_async_to_lds_b128(
      (__attribute__((address_space(1))) v4i*)g,
      (__attribute__((address_space(3))) v4i*)l, 0, 0);
}
#endif

__device__ __forceinline__ void wait_async0() {
#if __has_builtin(__builtin_amdgcn_s_wait_asynccnt)
  __builtin_amdgcn_s_wait_asynccnt(0);
#else
  asm volatile("s_wait_asynccnt 0x0" ::: "memory");
#endif
}

// ---------------------------------------------------------------------------
// Kernel 1: repack 3 weight matrices (f32 [K=1024][N=1024]) into bf16 B-operand
// fragment order:  flat = mat<<20 | (kb*64+nb)*512 | h2*256 | lane*8 | e7
//   k = kb*32 + 16*(lane>>4) + h2*8 + e7 ,  n = nb*16 + (lane&15)
// ---------------------------------------------------------------------------
__global__ void pack_w_kernel(const float* __restrict__ Wa,
                              const float* __restrict__ Wshift,
                              const float* __restrict__ Wscale,
                              __bf16* __restrict__ wb) {
  int idx = blockIdx.x * blockDim.x + threadIdx.x;   // < 3 * 1024 * 1024
  int mat    = idx >> 20;
  int rem    = idx & ((1 << 20) - 1);
  int frag   = rem >> 9;
  int within = rem & 511;
  int h2     = within >> 8;
  int lane   = (within >> 3) & 31;
  int e7     = within & 7;
  int kb     = frag >> 6;
  int nb     = frag & 63;
  int n      = nb * 16 + (lane & 15);
  int k      = kb * 32 + 16 * (lane >> 4) + h2 * 8 + e7;
  const float* W = (mat == 0) ? Wa : (mat == 1) ? Wshift : Wscale;
  wb[idx] = f2bf(W[(size_t)k * DD + n]);
}

// ---------------------------------------------------------------------------
// Kernel 2: causal Gaussian filter (16 taps capture sigma=2 to f32 precision).
// Writes x and filtered as row-major bf16 [8192][1024].
// ---------------------------------------------------------------------------
__global__ void filter_kernel(const float* __restrict__ x,
                              __bf16* __restrict__ xb,
                              __bf16* __restrict__ fb) {
  const int bt = blockIdx.x;            // b*S + t
  const int t  = bt & (SS - 1);
  const int d  = threadIdx.x * 4;
  const size_t base = (size_t)bt * DD + d;

  float4 x0 = *(const float4*)(x + base);
  float ax = 0.f, ay = 0.f, az = 0.f, aw = 0.f;
  float denom = 0.f;
  const int kmax = (t < (TAPS - 1)) ? t : (TAPS - 1);
  for (int k = 0; k <= kmax; ++k) {     // uniform across block
    float w = __expf(-(float)(k * k) * 0.125f);
    denom += w;
    float4 v = *(const float4*)(x + base - (size_t)(k * DD));
    ax += w * v.x; ay += w * v.y; az += w * v.z; aw += w * v.w;
  }
  float inv = 1.f / denom;
  v4bf xv, fv;
  xv[0] = f2bf(x0.x); xv[1] = f2bf(x0.y); xv[2] = f2bf(x0.z); xv[3] = f2bf(x0.w);
  fv[0] = f2bf(ax * inv); fv[1] = f2bf(ay * inv); fv[2] = f2bf(az * inv); fv[3] = f2bf(aw * inv);
  *(v4bf*)(xb + base) = xv;
  *(v4bf*)(fb + base) = fv;
}

// ---------------------------------------------------------------------------
// Stage the K-chunk `kb` into LDS buffer `nbuf`.
// Async path: DMA directly global->LDS (no VGPRs held).
// Fallback: load+store back-to-back (short live ranges, no spills).
// ---------------------------------------------------------------------------
__device__ __forceinline__ void stage_tile(const __bf16* __restrict__ stArow,
                                           const __bf16* __restrict__ wb,
                                           int nb0, int kb, char* nbuf,
                                           int tid, int stMat, int stRow, int stSwz) {
  const char* asrc = (const char*)(stArow + kb * KC);
  char* adst = nbuf + stMat * 8192 + stRow * 64;
#if HAVE_ASYNC_LDS
#pragma unroll
  for (int j = 0; j < 4; ++j)
    async_cp16(asrc + j * 16, adst + ((j ^ stSwz) * 16));
#pragma unroll
  for (int c3 = 0; c3 < 3; ++c3) {
    int c = tid + 256 * c3;             // 0..767
    int mat = c >> 8, off = c & 255;
    async_cp16(wb + ((size_t)mat << 20) + (size_t)(kb * 64 + nb0) * 512 + (size_t)off * 8,
               nbuf + LDS_B + c * 16);
  }
#else
  uint4 a[4];
#pragma unroll
  for (int j = 0; j < 4; ++j) a[j] = *(const uint4*)(asrc + j * 16);
#pragma unroll
  for (int j = 0; j < 4; ++j) *(uint4*)(adst + ((j ^ stSwz) * 16)) = a[j];
#pragma unroll
  for (int c3 = 0; c3 < 3; ++c3) {
    int c = tid + 256 * c3;
    int mat = c >> 8, off = c & 255;
    uint4 b = *(const uint4*)(wb + ((size_t)mat << 20) + (size_t)(kb * 64 + nb0) * 512 + (size_t)off * 8);
    *(uint4*)(nbuf + LDS_B + c * 16) = b;
  }
#endif
}

// ---------------------------------------------------------------------------
// Kernel 3: LDS-staged, double-buffered fused triple GEMM (bf16 WMMA),
// with async global->LDS prefetch overlapping the WMMA compute.
// ---------------------------------------------------------------------------
__global__ __launch_bounds__(256)
__attribute__((amdgpu_waves_per_eu(4)))
void fused_gemm_kernel(
    const __bf16* __restrict__ xb, const __bf16* __restrict__ fbuf,
    const __bf16* __restrict__ wb,
    const float* __restrict__ ba, const float* __restrict__ bshift,
    const float* __restrict__ bscale, float* __restrict__ out) {
  __shared__ uint4 lds4[2 * LDS_BUF / 16];
  char* lds = (char*)lds4;

  const int tid  = threadIdx.x;
  const int lane = tid & 31;
  const int wave = tid >> 5;
  const int wm = wave >> 1, wn = wave & 1;
  const int h  = lane >> 4;
  const int blockM = blockIdx.y * BM;
  const int blockN = blockIdx.x * BN;
  const int nb0 = blockN >> 4;          // first of 4 consecutive nb fragments

  // --- staging identities (uniform per thread across iterations) ---
  const int stMat = tid & 1;            // 0 = x, 1 = filtered
  const int stRow = tid >> 1;           // 0..127
  const __bf16* stArow = (stMat ? fbuf : xb) + (size_t)(blockM + stRow) * DD;
  const int stSwz = (stRow >> 2) & 3;   // A chunk swizzle for this row

  v8f zero = {};
  v8f accA[2][2], accH[2][2], accC[2][2];
#pragma unroll
  for (int mi = 0; mi < 2; ++mi)
#pragma unroll
    for (int ni = 0; ni < 2; ++ni) { accA[mi][ni] = zero; accH[mi][ni] = zero; accC[mi][ni] = zero; }

  // ---- prologue: stage kb=0 into buffer 0 ----
  stage_tile(stArow, wb, nb0, 0, lds, tid, stMat, stRow, stSwz);
#if HAVE_ASYNC_LDS
  wait_async0();
#endif
  __syncthreads();

  for (int kb = 0; kb < 32; ++kb) {
    char* buf = lds + (kb & 1) * LDS_BUF;

    // ---- prefetch kb+1 into the other buffer (overlaps WMMA compute) ----
    if (kb < 31) {
      char* nbuf = lds + ((kb + 1) & 1) * LDS_BUF;
      stage_tile(stArow, wb, nb0, kb + 1, nbuf, tid, stMat, stRow, stSwz);
    }

    // ---- load A fragments from LDS (chunk-swizzled rows) ----
    v16bf AX[2], AF[2];
#pragma unroll
    for (int mi = 0; mi < 2; ++mi) {
      int lr = wm * 32 + mi * 16 + (lane & 15);
      int sw = (lr >> 2) & 3;
      const v8bf* xr = (const v8bf*)(buf + LDS_A_X + lr * 64);
      const v8bf* fr = (const v8bf*)(buf + LDS_A_F + lr * 64);
      v8bf xlo = xr[(2 * h) ^ sw], xhi = xr[(2 * h + 1) ^ sw];
      v8bf flo = fr[(2 * h) ^ sw], fhi = fr[(2 * h + 1) ^ sw];
#pragma unroll
      for (int i = 0; i < 8; ++i) {
        AX[mi][i] = xlo[i]; AX[mi][i + 8] = xhi[i];
        AF[mi][i] = flo[i]; AF[mi][i + 8] = fhi[i];
      }
    }

    // ---- B fragments from LDS + 12 WMMAs ----
#pragma unroll
    for (int ni = 0; ni < 2; ++ni) {
      char* fa = buf + LDS_B + (0 * 4 + wn * 2 + ni) * 1024;
      char* fh = buf + LDS_B + (1 * 4 + wn * 2 + ni) * 1024;
      char* fc = buf + LDS_B + (2 * 4 + wn * 2 + ni) * 1024;
      v8bf alo = *(v8bf*)(fa + lane * 16), ahi = *(v8bf*)(fa + 512 + lane * 16);
      v8bf hlo = *(v8bf*)(fh + lane * 16), hhi = *(v8bf*)(fh + 512 + lane * 16);
      v8bf clo = *(v8bf*)(fc + lane * 16), chi = *(v8bf*)(fc + 512 + lane * 16);
      v16bf Bav, Bhv, Bcv;
#pragma unroll
      for (int i = 0; i < 8; ++i) {
        Bav[i] = alo[i]; Bav[i + 8] = ahi[i];
        Bhv[i] = hlo[i]; Bhv[i + 8] = hhi[i];
        Bcv[i] = clo[i]; Bcv[i + 8] = chi[i];
      }
#pragma unroll
      for (int mi = 0; mi < 2; ++mi) {
        accA[mi][ni] = __builtin_amdgcn_wmma_f32_16x16x32_bf16(false, AX[mi], false, Bav, (short)0, accA[mi][ni], false, false);
        accH[mi][ni] = __builtin_amdgcn_wmma_f32_16x16x32_bf16(false, AF[mi], false, Bhv, (short)0, accH[mi][ni], false, false);
        accC[mi][ni] = __builtin_amdgcn_wmma_f32_16x16x32_bf16(false, AF[mi], false, Bcv, (short)0, accC[mi][ni], false, false);
      }
    }

#if HAVE_ASYNC_LDS
    if (kb < 31) wait_async0();   // drain DMA into the next buffer
#endif
    __syncthreads();
  }

  // Epilogue: out = (f@Wscale + bscale) * relu(x@Wa + ba) + (f@Wshift + bshift)
#pragma unroll
  for (int ni = 0; ni < 2; ++ni) {
    const int col = blockN + wn * 32 + ni * 16 + (lane & 15);
    const float bav = ba[col], bhv = bshift[col], bcv = bscale[col];
#pragma unroll
    for (int mi = 0; mi < 2; ++mi) {
#pragma unroll
      for (int r = 0; r < 8; ++r) {
        const int mrow = blockM + wm * 32 + mi * 16 + r + 8 * h;
        float affine = accA[mi][ni][r] + bav;
        affine = fmaxf(affine, 0.0f);
        out[(size_t)mrow * DD + col] = (accC[mi][ni][r] + bcv) * affine + (accH[mi][ni][r] + bhv);
      }
    }
  }
}

// ---------------------------------------------------------------------------
extern "C" void kernel_launch(void* const* d_in, const int* in_sizes, int n_in,
                              void* d_out, int out_size, void* d_ws, size_t ws_size,
                              hipStream_t stream) {
  const float* x      = (const float*)d_in[0];
  const float* Wa     = (const float*)d_in[1];
  const float* ba     = (const float*)d_in[2];
  const float* Wshift = (const float*)d_in[3];
  const float* bshift = (const float*)d_in[4];
  const float* Wscale = (const float*)d_in[5];
  const float* bscale = (const float*)d_in[6];
  float* out = (float*)d_out;

  char* ws = (char*)d_ws;
  __bf16* xb   = (__bf16*)(ws);                   // 16 MB: x in bf16
  __bf16* fbuf = (__bf16*)(ws + (16u << 20));     // 16 MB: filtered in bf16
  __bf16* wb   = (__bf16*)(ws + (32u << 20));     // 6 MB: packed bf16 weights

  pack_w_kernel<<<(3u << 20) / 256, 256, 0, stream>>>(Wa, Wshift, Wscale, wb);
  filter_kernel<<<M_TOTAL, 256, 0, stream>>>(x, xb, fbuf);
  fused_gemm_kernel<<<dim3(DD / BN, M_TOTAL / BM), 256, 0, stream>>>(
      xb, fbuf, wb, ba, bshift, bscale, out);
}
